// MultiHeadAttention_mx_34909494182598
// MI455X (gfx1250) — matrix operand order repaired and checked by reading, not compiled, not run on hardware
//
#include <hip/hip_runtime.h>
#include <hip/hip_bf16.h>

// ---------------------------------------------------------------------------
// MI455X (gfx1250) multi-head attention: dual spatial/feature attention.
// f16 WMMA (v_wmma_f32_16x16x32_f16) tensor path, wave32, 8-wave blocks.
// Register-relay double buffering + async global->LDS staging (ASYNCcnt path)
// where the layout permits, single-stage K=64 GEMMs.
// ---------------------------------------------------------------------------

typedef __attribute__((ext_vector_type(16))) _Float16 v16h;
typedef __attribute__((ext_vector_type(8)))  float    v8f;
typedef __attribute__((ext_vector_type(2)))  _Float16 h2v;

#define DEVINL __device__ __forceinline__

static constexpr int BATCH = 16;
static constexpr int SEQ   = 1024;
static constexpr int DIMN  = 1024;
static constexpr int NH    = 16;
static constexpr int DH    = 64;
static constexpr int PITCH = 34;   // LDS row pitch (17 dwords, conflict-free)
static constexpr int QP    = 66;   // pitch for full-K=64 tiles (33 dwords, conflict-free)

#if __has_builtin(__builtin_amdgcn_global_load_async_to_lds_b32) && \
    __has_builtin(__builtin_amdgcn_s_wait_asynccnt)
#define HAS_ASYNC_LDS 1
#else
#define HAS_ASYNC_LDS 0
#endif

// Async global->LDS dword copy (ASYNCcnt-tracked); falls back to a plain copy.
// Builtin signature (from hipcc diagnostics): (AS1 int*, AS3 int*, imm, imm).
DEVINL void async_cp_b32(const void* g, void* l) {
#if HAS_ASYNC_LDS
    __builtin_amdgcn_global_load_async_to_lds_b32(
        (__attribute__((address_space(1))) int*)(unsigned long long)g,
        (__attribute__((address_space(3))) int*)(unsigned int)(unsigned long long)l,
        0, 0);
#else
    *(unsigned int*)l = *(const unsigned int*)g;
#endif
}
DEVINL void async_wait() {
#if HAS_ASYNC_LDS
    __builtin_amdgcn_s_wait_asynccnt(0);
#endif
}

DEVINL v8f v8f_zero() {
    v8f z;
#pragma unroll
    for (int i = 0; i < 8; ++i) z[i] = 0.f;
    return z;
}

// A fragment (16x32 f16, M x K). Lanes 0-15 -> M=0..15; K-pairs:
// VGPR0..7 = K{0..15}(+16 for lanes 16-31).
DEVINL v16h frag_a(const _Float16* t, int pitch) {
    const int lane = threadIdx.x & 31;
    const int g = lane >> 4, m = lane & 15;
    v16h f;
#pragma unroll
    for (int i = 0; i < 8; ++i) {
        const int kb = 2 * i + 16 * g;
        h2v p = *(const h2v*)(t + m * pitch + kb);
        f[2 * i] = p[0];
        f[2 * i + 1] = p[1];
    }
    return f;
}

// B fragment (32x16 f16, K x N) staged N-major (Bt[n][k]).
DEVINL v16h frag_b(const _Float16* t, int pitch) {
    const int lane = threadIdx.x & 31;
    const int g = lane >> 4, n = lane & 15;
    v16h f;
#pragma unroll
    for (int i = 0; i < 8; ++i) {
        const int kb = 2 * i + 16 * g;
        h2v p = *(const h2v*)(t + n * pitch + kb);
        f[2 * i] = p[0];
        f[2 * i + 1] = p[1];
    }
    return f;
}

DEVINL v8f wmma32(v16h a, v16h b, v8f c) {
    return __builtin_amdgcn_wmma_f32_16x16x32_f16(false, a, false, b, (short)0, c,
                                                  false, false);
}

// ---------------------------------------------------------------------------
__global__ void k_cvt_f16(const float* __restrict__ s, _Float16* __restrict__ d, int n) {
    int i = blockIdx.x * blockDim.x + threadIdx.x;
    const int stride = gridDim.x * blockDim.x;
    for (; i < n; i += stride) d[i] = (_Float16)s[i];
}

// ---------------------------------------------------------------------------
// Projection: X(16384x1024 fp32) @ W16 -> H f16 head-major (b,h,l,d).
// 128x128 tile, K-step 32, register-relay double buffering (fp32->f16 convert
// on A, transpose on B force the VGPR path).
// ---------------------------------------------------------------------------
__global__ __launch_bounds__(256) void k_gemm_proj(const float* __restrict__ X,
                                                   const _Float16* __restrict__ W,
                                                   _Float16* __restrict__ H) {
    __shared__ _Float16 sA[2][128 * PITCH];
    __shared__ _Float16 sB[2][128 * PITCH];
    const int tid = threadIdx.x, wave = tid >> 5;
    const int m0 = blockIdx.x * 128, n0 = blockIdx.y * 128;
    v8f acc[8];
#pragma unroll
    for (int t = 0; t < 8; ++t) acc[t] = v8f_zero();

    for (int idx = tid; idx < 128 * 32; idx += 256) {
        const int r = idx >> 5, c = idx & 31;
        sA[0][r * PITCH + c] = (_Float16)X[(size_t)(m0 + r) * DIMN + c];
    }
    for (int idx = tid; idx < 32 * 128; idx += 256) {
        const int k = idx >> 7, n = idx & 127;
        sB[0][n * PITCH + k] = W[(size_t)k * DIMN + n0 + n];
    }
    __syncthreads();

    for (int it = 0; it < 32; ++it) {
        const int buf = it & 1;
        const int k1 = (it + 1) * 32;
        float    ra[16];
        _Float16 rb[16];
        if (it < 31) {
#pragma unroll
            for (int j = 0; j < 16; ++j) {
                const int idx = tid + j * 256, r = idx >> 5, c = idx & 31;
                ra[j] = X[(size_t)(m0 + r) * DIMN + k1 + c];
            }
#pragma unroll
            for (int j = 0; j < 16; ++j) {
                const int idx = tid + j * 256, kk = idx >> 7, n = idx & 127;
                rb[j] = W[(size_t)(k1 + kk) * DIMN + n0 + n];
            }
        }
        const v16h a = frag_a(&sA[buf][wave * 16 * PITCH], PITCH);
#pragma unroll
        for (int nt = 0; nt < 8; ++nt)
            acc[nt] = wmma32(a, frag_b(&sB[buf][nt * 16 * PITCH], PITCH), acc[nt]);
        if (it < 31) {
#pragma unroll
            for (int j = 0; j < 16; ++j) {
                const int idx = tid + j * 256, r = idx >> 5, c = idx & 31;
                sA[buf ^ 1][r * PITCH + c] = (_Float16)ra[j];
            }
#pragma unroll
            for (int j = 0; j < 16; ++j) {
                const int idx = tid + j * 256, kk = idx >> 7, n = idx & 127;
                sB[buf ^ 1][n * PITCH + kk] = rb[j];
            }
        }
        __syncthreads();
    }

    const int lane = tid & 31, g = lane >> 4, nn = lane & 15;
#pragma unroll
    for (int nt = 0; nt < 8; ++nt)
#pragma unroll
        for (int r = 0; r < 8; ++r) {
            const int row = m0 + wave * 16 + r + 8 * g;   // b*SEQ + l
            const int col = n0 + nt * 16 + nn;            // h*DH + d
            const int b = row >> 10, l = row & 1023, h = col >> 6, d = col & 63;
            H[(((size_t)b * NH + h) * SEQ + l) * DH + d] = (_Float16)acc[nt][r];
        }
}

// ---------------------------------------------------------------------------
// Spatial scores: S = (Q/8)Kt per (b, even head hs). 128x128 tile, whole K=64
// staged once via async global->LDS (both tiles are k-contiguous f16).
// ---------------------------------------------------------------------------
__global__ __launch_bounds__(256) void k_gemm_qk(const _Float16* __restrict__ qh,
                                                 const _Float16* __restrict__ kh,
                                                 float* __restrict__ attn_s) {
    __shared__ _Float16 sA[128 * QP];
    __shared__ _Float16 sB[128 * QP];
    const int tid = threadIdx.x, wave = tid >> 5;
    const int pair = blockIdx.y;                 // b*8 + hs
    const int b = pair >> 3, hs = pair & 7, h = 2 * hs;
    const int m0 = (blockIdx.x >> 3) * 128, n0 = (blockIdx.x & 7) * 128;
    const _Float16* Q = qh + ((size_t)(b * NH + h)) * SEQ * DH;
    const _Float16* K = kh + ((size_t)(b * NH + h)) * SEQ * DH;

#pragma unroll
    for (int j = 0; j < 16; ++j) {               // 128 rows x 32 dwords
        const int idx = tid + j * 256, r = idx >> 5, cd = idx & 31;
        async_cp_b32(&Q[(size_t)(m0 + r) * DH + 2 * cd], &sA[r * QP + 2 * cd]);
    }
#pragma unroll
    for (int j = 0; j < 16; ++j) {
        const int idx = tid + j * 256, r = idx >> 5, cd = idx & 31;
        async_cp_b32(&K[(size_t)(n0 + r) * DH + 2 * cd], &sB[r * QP + 2 * cd]);
    }
    async_wait();
    __syncthreads();

    v8f acc[8];
#pragma unroll
    for (int t = 0; t < 8; ++t) acc[t] = v8f_zero();
#pragma unroll
    for (int ks = 0; ks < 2; ++ks) {
        const v16h a = frag_a(&sA[wave * 16 * QP + 32 * ks], QP);
#pragma unroll
        for (int nt = 0; nt < 8; ++nt)
            acc[nt] = wmma32(a, frag_b(&sB[nt * 16 * QP + 32 * ks], QP), acc[nt]);
    }

    const int lane = tid & 31, g = lane >> 4, nn = lane & 15;
    float* S = attn_s + (size_t)pair * SEQ * SEQ;
#pragma unroll
    for (int nt = 0; nt < 8; ++nt)
#pragma unroll
        for (int r = 0; r < 8; ++r) {
            const int row = m0 + wave * 16 + r + 8 * g;
            const int col = n0 + nt * 16 + nn;
            S[(size_t)row * SEQ + col] = acc[nt][r] * 0.125f;    // /sqrt(64)
        }
}

// ---------------------------------------------------------------------------
// Row softmax over 1024 cols, + mask_s, in place. One 256-thread block per row.
// ---------------------------------------------------------------------------
__global__ __launch_bounds__(256) void k_softmax_s(float* __restrict__ attn,
                                                   const float* __restrict__ mask) {
    __shared__ float red[256];
    const int tid = threadIdx.x;
    const size_t row = blockIdx.x;
    const int lrow = (int)(row & (SEQ - 1));
    float* base = attn + row * SEQ;
    const float* mrow = mask + (size_t)lrow * SEQ;
    float v[4], mx = -3.4e38f;
#pragma unroll
    for (int j = 0; j < 4; ++j) {
        const int c = tid + j * 256;
        v[j] = base[c] + mrow[c];
        mx = fmaxf(mx, v[j]);
    }
    red[tid] = mx; __syncthreads();
    for (int s = 128; s > 0; s >>= 1) {
        if (tid < s) red[tid] = fmaxf(red[tid], red[tid + s]);
        __syncthreads();
    }
    mx = red[0]; __syncthreads();
    float sum = 0.f;
#pragma unroll
    for (int j = 0; j < 4; ++j) { v[j] = __expf(v[j] - mx); sum += v[j]; }
    red[tid] = sum; __syncthreads();
    for (int s = 128; s > 0; s >>= 1) {
        if (tid < s) red[tid] += red[tid + s];
        __syncthreads();
    }
    const float inv = 1.f / red[0];
#pragma unroll
    for (int j = 0; j < 4; ++j) base[tid + j * 256] = v[j] * inv;
}

// ---------------------------------------------------------------------------
// ctx_even = attn_s @ V per (b,hs). 128x64 tile, K=1024, double-buffered
// register relay (A needs fp32->f16 convert, B needs transpose).
// ---------------------------------------------------------------------------
__global__ __launch_bounds__(256) void k_gemm_av(const float* __restrict__ attn_s,
                                                 const _Float16* __restrict__ vh,
                                                 _Float16* __restrict__ ctx) {
    __shared__ _Float16 sA[2][128 * PITCH];
    __shared__ _Float16 sB[2][64 * PITCH];
    const int tid = threadIdx.x, wave = tid >> 5;
    const int pair = blockIdx.y;
    const int b = pair >> 3, hs = pair & 7, h = 2 * hs;
    const int m0 = blockIdx.x * 128;
    const float* A = attn_s + (size_t)pair * SEQ * SEQ;
    const _Float16* V = vh + ((size_t)(b * NH + h)) * SEQ * DH;
    v8f acc[4];
#pragma unroll
    for (int t = 0; t < 4; ++t) acc[t] = v8f_zero();

    for (int idx = tid; idx < 128 * 32; idx += 256) {
        const int r = idx >> 5, c = idx & 31;
        sA[0][r * PITCH + c] = (_Float16)A[(size_t)(m0 + r) * SEQ + c];
    }
    for (int idx = tid; idx < 32 * 64; idx += 256) {             // Bt[d][l] = V[l][d]
        const int k = idx >> 6, n = idx & 63;
        sB[0][n * PITCH + k] = V[(size_t)k * DH + n];
    }
    __syncthreads();

    for (int it = 0; it < 32; ++it) {
        const int buf = it & 1;
        const int k1 = (it + 1) * 32;
        float    ra[16];
        _Float16 rb[8];
        if (it < 31) {
#pragma unroll
            for (int j = 0; j < 16; ++j) {
                const int idx = tid + j * 256, r = idx >> 5, c = idx & 31;
                ra[j] = A[(size_t)(m0 + r) * SEQ + k1 + c];
            }
#pragma unroll
            for (int j = 0; j < 8; ++j) {
                const int idx = tid + j * 256, kk = idx >> 6, n = idx & 63;
                rb[j] = V[(size_t)(k1 + kk) * DH + n];
            }
        }
        const v16h a = frag_a(&sA[buf][wave * 16 * PITCH], PITCH);
#pragma unroll
        for (int nt = 0; nt < 4; ++nt)
            acc[nt] = wmma32(a, frag_b(&sB[buf][nt * 16 * PITCH], PITCH), acc[nt]);
        if (it < 31) {
#pragma unroll
            for (int j = 0; j < 16; ++j) {
                const int idx = tid + j * 256, r = idx >> 5, c = idx & 31;
                sA[buf ^ 1][r * PITCH + c] = (_Float16)ra[j];
            }
#pragma unroll
            for (int j = 0; j < 8; ++j) {
                const int idx = tid + j * 256, kk = idx >> 6, n = idx & 63;
                sB[buf ^ 1][n * PITCH + kk] = rb[j];
            }
        }
        __syncthreads();
    }

    const int lane = tid & 31, g = lane >> 4, nn = lane & 15;
#pragma unroll
    for (int nt = 0; nt < 4; ++nt)
#pragma unroll
        for (int r = 0; r < 8; ++r) {
            const int row = m0 + wave * 16 + r + 8 * g;          // l
            const int col = nt * 16 + nn;                        // d
            ctx[((size_t)(b * SEQ + row)) * DIMN + hs * DH + col] = (_Float16)acc[nt][r];
        }
}

// ---------------------------------------------------------------------------
// Feature scores: S_f[n][m] = sum_l Q[l][n]K[l][m]/8 (odd heads), 64x64, K=1024.
// One block per (b,hf), double-buffered relay (transposed staging).
// ---------------------------------------------------------------------------
__global__ __launch_bounds__(256) void k_gemm_ff_s(const _Float16* __restrict__ qh,
                                                   const _Float16* __restrict__ kh,
                                                   float* __restrict__ attn_f) {
    __shared__ _Float16 sA[2][64 * PITCH];
    __shared__ _Float16 sB[2][64 * PITCH];
    const int tid = threadIdx.x, wave = tid >> 5;
    const int pair = blockIdx.x;
    const int b = pair >> 3, hf = pair & 7, h = 2 * hf + 1;
    const _Float16* Q = qh + ((size_t)(b * NH + h)) * SEQ * DH;
    const _Float16* K = kh + ((size_t)(b * NH + h)) * SEQ * DH;
    const int mt = wave >> 1, ntb = wave & 1;
    v8f acc0 = v8f_zero(), acc1 = v8f_zero();

    for (int idx = tid; idx < 32 * 64; idx += 256) {
        const int k = idx >> 6, m = idx & 63;
        sA[0][m * PITCH + k] = Q[(size_t)k * DH + m];
        sB[0][m * PITCH + k] = K[(size_t)k * DH + m];
    }
    __syncthreads();

    for (int it = 0; it < 32; ++it) {
        const int buf = it & 1;
        const int k1 = (it + 1) * 32;
        _Float16 rq[8], rk[8];
        if (it < 31) {
#pragma unroll
            for (int j = 0; j < 8; ++j) {
                const int idx = tid + j * 256, kk = idx >> 6, m = idx & 63;
                rq[j] = Q[(size_t)(k1 + kk) * DH + m];
                rk[j] = K[(size_t)(k1 + kk) * DH + m];
            }
        }
        const v16h a = frag_a(&sA[buf][mt * 16 * PITCH], PITCH);
        acc0 = wmma32(a, frag_b(&sB[buf][ntb * 16 * PITCH], PITCH), acc0);
        acc1 = wmma32(a, frag_b(&sB[buf][(ntb + 2) * 16 * PITCH], PITCH), acc1);
        if (it < 31) {
#pragma unroll
            for (int j = 0; j < 8; ++j) {
                const int idx = tid + j * 256, kk = idx >> 6, m = idx & 63;
                sA[buf ^ 1][m * PITCH + kk] = rq[j];
                sB[buf ^ 1][m * PITCH + kk] = rk[j];
            }
        }
        __syncthreads();
    }

    const int lane = tid & 31, g = lane >> 4, nn = lane & 15;
    float* S = attn_f + (size_t)pair * DH * DH;
#pragma unroll
    for (int r = 0; r < 8; ++r) {
        const int row = mt * 16 + r + 8 * g;
        S[(size_t)row * DH + ntb * 16 + nn]       = acc0[r] * 0.125f;
        S[(size_t)row * DH + (ntb + 2) * 16 + nn] = acc1[r] * 0.125f;
    }
}

// ---------------------------------------------------------------------------
// Feature softmax: rows of 64, + mask_f. One 64-thread block per row.
// ---------------------------------------------------------------------------
__global__ __launch_bounds__(64) void k_softmax_f(float* __restrict__ attn,
                                                  const float* __restrict__ mask) {
    __shared__ float red[64];
    const int tid = threadIdx.x;
    const size_t row = blockIdx.x;
    const int lrow = (int)(row & (DH - 1));
    float* base = attn + row * DH;
    float v = base[tid] + mask[(size_t)lrow * DH + tid];
    red[tid] = v; __syncthreads();
    for (int s = 32; s > 0; s >>= 1) {
        if (tid < s) red[tid] = fmaxf(red[tid], red[tid + s]);
        __syncthreads();
    }
    const float mx = red[0]; __syncthreads();
    v = __expf(v - mx);
    red[tid] = v; __syncthreads();
    for (int s = 32; s > 0; s >>= 1) {
        if (tid < s) red[tid] += red[tid + s];
        __syncthreads();
    }
    base[tid] = v / red[0];
}

// ---------------------------------------------------------------------------
// ctx_odd[l][n] = sum_m V[l][m]*attn_f[n][m] (== Vh @ attn_f^T), K=64.
// Whole-K single stage: V tile via async copy, attn_f converted in place.
// ---------------------------------------------------------------------------
__global__ __launch_bounds__(256) void k_gemm_ff_o(const _Float16* __restrict__ vh,
                                                   const float* __restrict__ attn_f,
                                                   _Float16* __restrict__ ctx) {
    __shared__ _Float16 sA[128 * QP];
    __shared__ _Float16 sB[64 * QP];      // Bt[n][m] = attn_f[n][m]
    const int tid = threadIdx.x, wave = tid >> 5;
    const int pair = blockIdx.y;
    const int b = pair >> 3, hf = pair & 7, h = 2 * hf + 1;
    const int m0 = blockIdx.x * 128;
    const _Float16* V = vh + ((size_t)(b * NH + h)) * SEQ * DH;
    const float* Af = attn_f + (size_t)pair * DH * DH;

#pragma unroll
    for (int j = 0; j < 8; ++j) {                // 64 rows x 32 dwords
        const int idx = tid + j * 256, r = idx >> 5, cd = idx & 31;
        async_cp_b32(&V[(size_t)(m0 + r) * DH + 2 * cd], &sA[r * QP + 2 * cd]);
    }
    for (int idx = tid; idx < 64 * 64; idx += 256) {
        const int n = idx >> 6, m = idx & 63;
        sB[n * QP + m] = (_Float16)Af[(size_t)n * DH + m];
    }
    async_wait();
    __syncthreads();

    v8f acc[4];
#pragma unroll
    for (int t = 0; t < 4; ++t) acc[t] = v8f_zero();
#pragma unroll
    for (int ks = 0; ks < 2; ++ks) {
        const v16h a = frag_a(&sA[wave * 16 * QP + 32 * ks], QP);
#pragma unroll
        for (int nt = 0; nt < 4; ++nt)
            acc[nt] = wmma32(a, frag_b(&sB[nt * 16 * QP + 32 * ks], QP), acc[nt]);
    }

    const int lane = tid & 31, g = lane >> 4, nn = lane & 15;
#pragma unroll
    for (int nt = 0; nt < 4; ++nt)
#pragma unroll
        for (int r = 0; r < 8; ++r) {
            const int row = m0 + wave * 16 + r + 8 * g;          // l
            const int col = nt * 16 + nn;                        // n
            ctx[((size_t)(b * SEQ + row)) * DIMN + (8 + hf) * DH + col] = (_Float16)acc[nt][r];
        }
}

// ---------------------------------------------------------------------------
// Output projection: ctx(16384x1024 f16) @ Wo16 -> out fp32. Double buffered:
// A tile via async global->LDS ping-pong, B (transpose) via register relay.
// ---------------------------------------------------------------------------
__global__ __launch_bounds__(256) void k_gemm_out(const _Float16* __restrict__ X,
                                                  const _Float16* __restrict__ W,
                                                  float* __restrict__ out) {
    __shared__ _Float16 sA[2][128 * PITCH];
    __shared__ _Float16 sB[2][128 * PITCH];
    const int tid = threadIdx.x, wave = tid >> 5;
    const int m0 = blockIdx.x * 128, n0 = blockIdx.y * 128;
    v8f acc[8];
#pragma unroll
    for (int t = 0; t < 8; ++t) acc[t] = v8f_zero();

#pragma unroll
    for (int j = 0; j < 8; ++j) {                // 128 rows x 16 dwords
        const int idx = tid + j * 256, r = idx >> 4, cd = idx & 15;
        async_cp_b32(&X[(size_t)(m0 + r) * DIMN + 2 * cd], &sA[0][r * PITCH + 2 * cd]);
    }
    for (int idx = tid; idx < 32 * 128; idx += 256) {
        const int k = idx >> 7, n = idx & 127;
        sB[0][n * PITCH + k] = W[(size_t)k * DIMN + n0 + n];
    }
    async_wait();
    __syncthreads();

    for (int it = 0; it < 32; ++it) {
        const int buf = it & 1;
        const int k1 = (it + 1) * 32;
        _Float16 rb[16];
        if (it < 31) {
#pragma unroll
            for (int j = 0; j < 8; ++j) {
                const int idx = tid + j * 256, r = idx >> 4, cd = idx & 15;
                async_cp_b32(&X[(size_t)(m0 + r) * DIMN + k1 + 2 * cd],
                             &sA[buf ^ 1][r * PITCH + 2 * cd]);
            }
#pragma unroll
            for (int j = 0; j < 16; ++j) {
                const int idx = tid + j * 256, kk = idx >> 7, n = idx & 127;
                rb[j] = W[(size_t)(k1 + kk) * DIMN + n0 + n];
            }
        }
        const v16h a = frag_a(&sA[buf][wave * 16 * PITCH], PITCH);
#pragma unroll
        for (int nt = 0; nt < 8; ++nt)
            acc[nt] = wmma32(a, frag_b(&sB[buf][nt * 16 * PITCH], PITCH), acc[nt]);
        if (it < 31) {
#pragma unroll
            for (int j = 0; j < 16; ++j) {
                const int idx = tid + j * 256, kk = idx >> 7, n = idx & 127;
                sB[buf ^ 1][n * PITCH + kk] = rb[j];
            }
            async_wait();
        }
        __syncthreads();
    }

    const int lane = tid & 31, g = lane >> 4, nn = lane & 15;
#pragma unroll
    for (int nt = 0; nt < 8; ++nt)
#pragma unroll
        for (int r = 0; r < 8; ++r) {
            const int row = m0 + wave * 16 + r + 8 * g;
            const int col = n0 + nt * 16 + nn;
            out[(size_t)row * DIMN + col] = acc[nt][r];
        }
}

// ---------------------------------------------------------------------------
extern "C" void kernel_launch(void* const* d_in, const int* in_sizes, int n_in,
                              void* d_out, int out_size, void* d_ws, size_t ws_size,
                              hipStream_t stream) {
    (void)in_sizes; (void)n_in; (void)out_size; (void)ws_size;

    const float* q      = (const float*)d_in[0];
    const float* k      = (const float*)d_in[1];
    const float* v      = (const float*)d_in[2];
    const float* mask_s = (const float*)d_in[3];
    const float* mask_f = (const float*)d_in[4];
    const float* Wq     = (const float*)d_in[5];
    const float* Wk     = (const float*)d_in[6];
    const float* Wv     = (const float*)d_in[7];
    const float* Wo     = (const float*)d_in[8];

    float* out    = (float*)d_out;                                   // (16,1024,1024)
    float* attn_s = out + (size_t)BATCH * SEQ * DIMN;                // (16,8,1024,1024)
    float* attn_f = attn_s + (size_t)BATCH * 8 * SEQ * SEQ;          // (16,8,64,64)

    char* w = (char*)d_ws;
    auto take = [&](size_t bytes) { void* p = (void*)w; w += (bytes + 255) & ~(size_t)255; return p; };
    const size_t WB = (size_t)DIMN * DIMN * sizeof(_Float16);           // 2 MB
    const size_t HB = (size_t)BATCH * NH * SEQ * DH * sizeof(_Float16); // 32 MB
    _Float16* wq16  = (_Float16*)take(WB);
    _Float16* wk16  = (_Float16*)take(WB);
    _Float16* wv16  = (_Float16*)take(WB);
    _Float16* wo16  = (_Float16*)take(WB);
    _Float16* qh16  = (_Float16*)take(HB);
    _Float16* kh16  = (_Float16*)take(HB);
    _Float16* vh16  = (_Float16*)take(HB);
    _Float16* ctx16 = (_Float16*)take((size_t)BATCH * SEQ * DIMN * sizeof(_Float16));

    const int NW = DIMN * DIMN;
    k_cvt_f16<<<1024, 256, 0, stream>>>(Wq, wq16, NW);
    k_cvt_f16<<<1024, 256, 0, stream>>>(Wk, wk16, NW);
    k_cvt_f16<<<1024, 256, 0, stream>>>(Wv, wv16, NW);
    k_cvt_f16<<<1024, 256, 0, stream>>>(Wo, wo16, NW);

    const dim3 gProj(BATCH * SEQ / 128, DIMN / 128);                 // (128, 8)
    k_gemm_proj<<<gProj, 256, 0, stream>>>(q, wq16, qh16);
    k_gemm_proj<<<gProj, 256, 0, stream>>>(k, wk16, kh16);
    k_gemm_proj<<<gProj, 256, 0, stream>>>(v, wv16, vh16);

    k_gemm_qk<<<dim3(64, BATCH * 8), 256, 0, stream>>>(qh16, kh16, attn_s);
    k_softmax_s<<<BATCH * 8 * SEQ, 256, 0, stream>>>(attn_s, mask_s);
    k_gemm_av<<<dim3(SEQ / 128, BATCH * 8), 256, 0, stream>>>(attn_s, vh16, ctx16);

    k_gemm_ff_s<<<BATCH * 8, 256, 0, stream>>>(qh16, kh16, attn_f);
    k_softmax_f<<<BATCH * 8 * DH, 64, 0, stream>>>(attn_f, mask_f);
    k_gemm_ff_o<<<dim3(SEQ / 128, BATCH * 8), 256, 0, stream>>>(vh16, attn_f, ctx16);

    k_gemm_out<<<dim3(BATCH * SEQ / 128, DIMN / 128), 256, 0, stream>>>(ctx16, wo16, out);
}
